// GCN_62448824484016
// MI455X (gfx1250) — matrix operand order repaired and checked by reading, not compiled, not run on hardware
//
#include <hip/hip_runtime.h>

// ---- vector types (own ext_vector typedefs: trivially-copyable, union-safe) ----
typedef __attribute__((ext_vector_type(16))) _Float16 v16h;
typedef __attribute__((ext_vector_type(8)))  float    v8f;
typedef __attribute__((ext_vector_type(4)))  float    f32x4;

#define D_IN  512
#define D_OUT 64

// ---------------------------------------------------------------------------
// packed f16 atomic add (CDNA5 GLOBAL_ATOMIC_PK_ADD_F16, no-return form)
// ---------------------------------------------------------------------------
__device__ __forceinline__ void atomic_pk_add_f16(_Float16* addr, unsigned pk2) {
    asm volatile("global_atomic_pk_add_f16 %0, %1, off"
                 :: "v"((unsigned long long)(__SIZE_TYPE__)addr), "v"(pk2)
                 : "memory");
}

// ---------------------------------------------------------------------------
// fp32 -> f16 convert, 8 elements/thread (b128 in, b128 out)
// ---------------------------------------------------------------------------
__global__ void k_cvt8(const float* __restrict__ in, _Float16* __restrict__ out,
                       long long n8) {
    long long i = blockIdx.x * (long long)blockDim.x + threadIdx.x;
    if (i >= n8) return;
    const f32x4* p = (const f32x4*)in + i * 2;
    f32x4 a = p[0], b = p[1];
    union { _Float16 h[8]; f32x4 q; } u;
    u.h[0] = (_Float16)a.x; u.h[1] = (_Float16)a.y;
    u.h[2] = (_Float16)a.z; u.h[3] = (_Float16)a.w;
    u.h[4] = (_Float16)b.x; u.h[5] = (_Float16)b.y;
    u.h[6] = (_Float16)b.z; u.h[7] = (_Float16)b.w;
    ((f32x4*)out)[i] = u.q;
}

// ---------------------------------------------------------------------------
// zero-fill, 16B granularity
// ---------------------------------------------------------------------------
__global__ void k_zero(f32x4* __restrict__ p, long long n16) {
    long long i = blockIdx.x * (long long)blockDim.x + threadIdx.x;
    f32x4 z = {0.f, 0.f, 0.f, 0.f};
    if (i < n16) p[i] = z;
}

// ---------------------------------------------------------------------------
// f16 edge scatter-add: acc[dst[e]][:] += X[src[e]][:]
// one thread = one edge x 16 halves (2x b128 gather + 8x global_atomic_pk_add_f16)
// ---------------------------------------------------------------------------
__global__ void k_scatter_pk(const _Float16* __restrict__ X,
                             const int* __restrict__ src,
                             const int* __restrict__ dst,
                             _Float16* __restrict__ acc, int E) {
    long long tid = blockIdx.x * (long long)blockDim.x + threadIdx.x;
    int e = (int)(tid >> 5);
    if (e >= E) return;
    int c = ((int)tid & 31) << 4;  // 16-half chunk
    const f32x4* xp = (const f32x4*)(X + (long long)src[e] * D_IN + c);
    _Float16*    op = acc + (long long)dst[e] * D_IN + c;
    f32x4 v0 = xp[0], v1 = xp[1];
    union U { f32x4 q; unsigned u[4]; };
    U a; a.q = v0;
    U b; b.q = v1;
#pragma unroll
    for (int i = 0; i < 4; ++i) atomic_pk_add_f16(op + i * 2, a.u[i]);
#pragma unroll
    for (int i = 0; i < 4; ++i) atomic_pk_add_f16(op + 8 + i * 2, b.u[i]);
}

// ---------------------------------------------------------------------------
// GEMM1: C[n,e] = f16( relu( sum_d A[n,d] * W1[e,d] ) ),  A/W1 f16, WMMA
// block = 8 waves; block owns rows [16*bx, 16*bx+16), wave w owns cols [64w,64w+64)
// ---------------------------------------------------------------------------
__global__ __launch_bounds__(256) void k_gemm1(const _Float16* __restrict__ A,
                                               const _Float16* __restrict__ Bh,
                                               _Float16* __restrict__ C) {
    const int rBase = blockIdx.x * 16;
    const int wv   = threadIdx.x >> 5;   // 0..7  -> column group
    const int lane = threadIdx.x & 31;
    const int m    = lane & 15;
    const int hi   = lane >> 4;          // 0/1

    __shared__ _Float16 As[16 * D_IN];   // 16 KB staged A tile

    // cooperative b128 copy of the 16x512 f16 A tile into LDS
    {
        const f32x4* gp = (const f32x4*)(A + (long long)rBase * D_IN);
        f32x4*       lp = (f32x4*)As;
        for (int i = threadIdx.x; i < 1024; i += 256) lp[i] = gp[i];
    }
    __syncthreads();

    const int cBase = wv * 64;
    v8f acc[4] = {};

    // A fragment (ISA 7.12.2 16-bit A layout): lanes0-15 K {0..7,16..23}, lanes16-31 K {8..15,24..31}
    const _Float16* as  = As + m * D_IN + hi * 8;
    // B fragment: lane holds column (cBase+16j+m), K {0..7,16..23} (lanes0-15) / {8..15,24..31} (lanes16-31)
    const _Float16* bp0 = Bh + (long long)(cBase + m) * D_IN + hi * 8;

#pragma unroll 4
    for (int kk = 0; kk < D_IN; kk += 32) {
        union { v16h v; f32x4 q[2]; } af;
        af.q[0] = *(const f32x4*)(as + kk);        // ds_load_b128
        af.q[1] = *(const f32x4*)(as + kk + 16);
#pragma unroll
        for (int j = 0; j < 4; ++j) {
            union { v16h v; f32x4 q[2]; } bf;
            const _Float16* bp = bp0 + (long long)j * 16 * D_IN + kk;
            bf.q[0] = *(const f32x4*)bp;
            bf.q[1] = *(const f32x4*)(bp + 16);
            acc[j] = __builtin_amdgcn_wmma_f32_16x16x32_f16(
                false, af.v, false, bf.v, (short)0, acc[j], false, false);
        }
    }

    // C layout: VGPR i -> row i (lanes0-15) / row i+8 (lanes16-31), col = m
#pragma unroll
    for (int j = 0; j < 4; ++j)
#pragma unroll
        for (int i = 0; i < 8; ++i) {
            int row = rBase + i + hi * 8;
            int col = cBase + j * 16 + m;
            C[(long long)row * D_IN + col] = (_Float16)fmaxf(acc[j][i], 0.0f);
        }
}

// ---------------------------------------------------------------------------
// GEMM2 + log_softmax fused: out[n,:] = log_softmax( sum_d A[n,d] * W2[o,d] )
// one wave per 16-row tile; full 16x64 output in 4 WMMA accumulators; A is f16
// ---------------------------------------------------------------------------
__global__ __launch_bounds__(32) void k_gemm2_lsm(const _Float16* __restrict__ A,
                                                  const _Float16* __restrict__ Bh,
                                                  float* __restrict__ out) {
    const int rBase = blockIdx.x * 16;
    const int lane  = threadIdx.x;
    const int m     = lane & 15;
    const int hi    = lane >> 4;

    __shared__ float tile[16][68];

    v8f acc[4] = {};
    const _Float16* ap  = A  + (long long)(rBase + m) * D_IN + hi * 8;
    const _Float16* bp0 = Bh + (long long)m * D_IN + hi * 8;

#pragma unroll 4
    for (int kk = 0; kk < D_IN; kk += 32) {
        union { v16h v; f32x4 q[2]; } af;
        af.q[0] = *(const f32x4*)(ap + kk);
        af.q[1] = *(const f32x4*)(ap + kk + 16);
#pragma unroll
        for (int j = 0; j < 4; ++j) {
            union { v16h v; f32x4 q[2]; } bf;
            const _Float16* bp = bp0 + (long long)j * 16 * D_IN + kk;
            bf.q[0] = *(const f32x4*)bp;
            bf.q[1] = *(const f32x4*)(bp + 16);
            acc[j] = __builtin_amdgcn_wmma_f32_16x16x32_f16(
                false, af.v, false, bf.v, (short)0, acc[j], false, false);
        }
    }

#pragma unroll
    for (int j = 0; j < 4; ++j)
#pragma unroll
        for (int i = 0; i < 8; ++i)
            tile[i + hi * 8][j * 16 + m] = acc[j][i];
    __syncthreads();

    if (lane < 16) {
        int row = rBase + lane;
        float mx = -3.4e38f;
#pragma unroll 8
        for (int j = 0; j < D_OUT; ++j) mx = fmaxf(mx, tile[lane][j]);
        float s = 0.0f;
#pragma unroll 8
        for (int j = 0; j < D_OUT; ++j) s += __expf(tile[lane][j] - mx);
        float l = __logf(s);
        float* op = out + (long long)row * D_OUT;
#pragma unroll 8
        for (int j = 0; j < D_OUT; ++j) op[j] = tile[lane][j] - mx - l;
    }
}

// ---------------------------------------------------------------------------
// launch
// ---------------------------------------------------------------------------
extern "C" void kernel_launch(void* const* d_in, const int* in_sizes, int n_in,
                              void* d_out, int out_size, void* d_ws, size_t ws_size,
                              hipStream_t stream) {
    const float* x  = (const float*)d_in[0];
    const int*   ei = (const int*)d_in[1];   // [2, E]: row0 = src, row1 = dst
    const float* W1 = (const float*)d_in[2];
    const float* W2 = (const float*)d_in[3];
    float* out = (float*)d_out;

    const int N = in_sizes[0] / D_IN;        // 20000
    const int E = in_sizes[1] / 2;           // 160000
    const size_t nh = (size_t)N * D_IN;      // halves per node-feature matrix

    // workspace layout (all f16): xh | h1h (reused as h2agg) | h2h | W1h | W2h
    _Float16* xh  = (_Float16*)d_ws;
    _Float16* h1h = xh + nh;
    _Float16* h2h = h1h + nh;
    _Float16* W1h = h2h + nh;
    _Float16* W2h = W1h + (size_t)D_IN * D_IN;

    // fp32 -> f16 converts (x is 61MB of traffic; weights are L2-resident after)
    const long long xn8 = (long long)(nh / 8);
    k_cvt8<<<(int)((xn8 + 255) / 256), 256, 0, stream>>>(x, xh, xn8);
    k_cvt8<<<(D_IN * D_IN / 8 + 255) / 256, 256, 0, stream>>>(W1, W1h, D_IN * D_IN / 8);
    k_cvt8<<<(D_OUT * D_IN / 8 + 255) / 256, 256, 0, stream>>>(W2, W2h, D_OUT * D_IN / 8);

    const long long n16 = (long long)(nh * 2 / 16);             // 16B chunks
    const int zgrid = (int)((n16 + 255) / 256);
    const int sgrid = (int)(((long long)E * 32 + 255) / 256);

    // propagate 1: h1 = segment_sum(x[src] -> dst)   (pk f16 atomics)
    k_zero<<<zgrid, 256, 0, stream>>>((f32x4*)h1h, n16);
    k_scatter_pk<<<sgrid, 256, 0, stream>>>(xh, ei, ei + E, h1h, E);

    // lin1 + relu (WMMA)
    k_gemm1<<<N / 16, 256, 0, stream>>>(h1h, W1h, h2h);

    // propagate 2: h2agg = segment_sum(h2[src] -> dst)   (reuse h1h buffer)
    k_zero<<<zgrid, 256, 0, stream>>>((f32x4*)h1h, n16);
    k_scatter_pk<<<sgrid, 256, 0, stream>>>(h2h, ei, ei + E, h1h, E);

    // lin2 + log_softmax (WMMA, fused)
    k_gemm2_lsm<<<N / 16, 32, 0, stream>>>(h1h, W2h, out);
}
